// MambaBlock_17583596109964
// MI455X (gfx1250) — compile-verified, matrix operand order checked by reading, not run on hardware
//
#include <hip/hip_runtime.h>
#include <math.h>

// ---- problem constants (match reference) ----
#define Bb   2
#define Ll   2048
#define Dd   768
#define DIc  1536
#define Nn   16
#define Kc   4
#define DTR  48
#define DTRP 64              // dt-rank padded to a multiple of 32 for WMMA K
#define ROWS (Bb*Ll)         // 4096
#define XPROJ (DTR + 2*Nn)   // 80

typedef __bf16 bf16;
typedef __bf16 bfx8  __attribute__((ext_vector_type(8)));
typedef __bf16 bfx16 __attribute__((ext_vector_type(16)));
typedef float  fx8   __attribute__((ext_vector_type(8)));

__device__ __forceinline__ bf16 f2bf(float f) {
    unsigned u = __builtin_bit_cast(unsigned, f);
    unsigned r = (u + 0x7FFFu + ((u >> 16) & 1u)) >> 16;   // round-to-nearest-even
    unsigned short s = (unsigned short)r;
    return __builtin_bit_cast(bf16, s);
}

// One 16x32 bf16 WMMA fragment for this lane: per the ISA VGPR layout it is
// exactly two contiguous 8-halfword runs of the K axis (16B each):
//   elems 0..7  = row[kb + 8*half .. +7]
//   elems 8..15 = row[kb + 16 + 8*half .. +7]
// All GEMM K values are multiples of 32 (dt path padded), so no guards:
// two unconditional global_load_b128 per fragment.
__device__ __forceinline__ bfx16 load_frag(const bf16* __restrict__ row, int kb, int hf) {
    const bfx8 lo = *(const bfx8*)(row + kb + hf * 8);
    const bfx8 hi = *(const bfx8*)(row + kb + 16 + hf * 8);
    return __builtin_shufflevector(lo, hi, 0, 1, 2, 3, 4, 5, 6, 7,
                                   8, 9, 10, 11, 12, 13, 14, 15);
}

// ---------------- LayerNorm -> bf16 ----------------
__global__ __launch_bounds__(256) void layernorm_kernel(
    const float* __restrict__ x, const float* __restrict__ g,
    const float* __restrict__ be, bf16* __restrict__ xn)
{
    __shared__ float s1[256], s2[256];
    const int row = blockIdx.x;
    const float* xr = x + (size_t)row * Dd;
    float a = 0.f, b = 0.f, v[3];
#pragma unroll
    for (int k = 0; k < 3; ++k) {
        v[k] = xr[threadIdx.x + k * 256];
        a += v[k]; b += v[k] * v[k];
    }
    s1[threadIdx.x] = a; s2[threadIdx.x] = b;
    __syncthreads();
    for (int off = 128; off > 0; off >>= 1) {
        if (threadIdx.x < off) {
            s1[threadIdx.x] += s1[threadIdx.x + off];
            s2[threadIdx.x] += s2[threadIdx.x + off];
        }
        __syncthreads();
    }
    const float mean = s1[0] * (1.f / Dd);
    const float var  = s2[0] * (1.f / Dd) - mean * mean;
    const float rstd = rsqrtf(var + 1e-5f);
#pragma unroll
    for (int k = 0; k < 3; ++k) {
        const int i = threadIdx.x + k * 256;
        xn[(size_t)row * Dd + i] = f2bf((v[k] - mean) * rstd * g[i] + be[i]);
    }
}

// ---------------- fp32 -> bf16 convert ----------------
__global__ __launch_bounds__(256) void convert_bf16_kernel(
    const float* __restrict__ src, bf16* __restrict__ dst, int n)
{
    int i = blockIdx.x * 256 + threadIdx.x;
    if (i < n) dst[i] = f2bf(src[i]);
}

// fp32 (rows x kin) -> bf16 (rows x kout) with zero pad in [kin, kout)
__global__ __launch_bounds__(256) void convert_pad_bf16_kernel(
    const float* __restrict__ src, bf16* __restrict__ dst,
    int rows, int kin, int kout)
{
    int i = blockIdx.x * 256 + threadIdx.x;
    if (i >= rows * kout) return;
    const int r = i / kout, k = i % kout;
    dst[i] = (k < kin) ? f2bf(src[r * kin + k]) : f2bf(0.f);
}

// ---------------- bf16 WMMA GEMM: out = A(MxK) * W(NxK)^T ----------------
// 256 threads = 8 waves; each wave computes a 32(M) x 64(N) strip:
// 2 m-tiles x 4 n-tiles = 8 accumulators, B fragments reused across m-tiles.
// Block tile = 256(M) x 64(N). K must be a multiple of 32 (guaranteed).
// FULLN=true compiles out the n-tile bounds guard (N multiple of 64).
// MODE: 1 split-store (out0 | out1), 2 f32 + padded-bf16 copy of dt cols,
//       3 softplus(v + bias[n]), 4 residual add from `resid`, 0 plain.
template <int MODE, bool FULLN>
__global__ __launch_bounds__(256) void wmma_gemm_kernel(
    const bf16* __restrict__ A, const bf16* __restrict__ W,
    int N, int K,
    float* __restrict__ out0, float* __restrict__ out1,
    bf16* __restrict__ outb, const float* __restrict__ bias,
    const float* __restrict__ resid)
{
    const int lane  = threadIdx.x & 31;
    const int wv    = threadIdx.x >> 5;   // wave in block: 0..7
    const int hf    = lane >> 4;          // lane-half 0/1
    const int l16   = lane & 15;
    const int mBase = blockIdx.x * 256 + wv * 32;
    const int nBase = blockIdx.y * 64;

    fx8 acc[2][4];
#pragma unroll
    for (int mi = 0; mi < 2; ++mi)
#pragma unroll
        for (int t = 0; t < 4; ++t)
#pragma unroll
            for (int i = 0; i < 8; ++i) acc[mi][t][i] = 0.f;

    const bf16* arow0 = A + (size_t)(mBase + l16) * K;
    const bf16* arow1 = A + (size_t)(mBase + 16 + l16) * K;

    for (int kb = 0; kb < K; kb += 32) {
        const bfx16 a0 = load_frag(arow0, kb, hf);
        const bfx16 a1 = load_frag(arow1, kb, hf);
#pragma unroll
        for (int t = 0; t < 4; ++t) {
            const int n0 = nBase + t * 16;
            if (FULLN || n0 < N) {                     // uniform; gone when FULLN
                const bf16* wrow = W + (size_t)(n0 + l16) * K;
                const bfx16 bfr = load_frag(wrow, kb, hf);
                acc[0][t] = __builtin_amdgcn_wmma_f32_16x16x32_bf16(
                    false, a0, false, bfr, (short)0, acc[0][t], false, false);
                acc[1][t] = __builtin_amdgcn_wmma_f32_16x16x32_bf16(
                    false, a1, false, bfr, (short)0, acc[1][t], false, false);
            }
        }
    }

    // C/D layout: lane -> n = lane%16; VGPR j -> m = j + 8*(lane/16)
#pragma unroll
    for (int mi = 0; mi < 2; ++mi) {
#pragma unroll
        for (int t = 0; t < 4; ++t) {
            const int n0 = nBase + t * 16;
            if (!FULLN && n0 >= N) continue;
            const int gn = n0 + l16;
#pragma unroll
            for (int j = 0; j < 8; ++j) {
                const int gm = mBase + mi * 16 + hf * 8 + j;
                float v = acc[mi][t][j];
                if (MODE == 1) {
                    const int nh = N >> 1;
                    if (gn < nh) out0[(size_t)gm * nh + gn] = v;
                    else         out1[(size_t)gm * nh + (gn - nh)] = v;
                } else if (MODE == 2) {
                    out0[(size_t)gm * N + gn] = v;
                    if (gn < DTRP)   // write padded dt-lowrank row (zeros in pad)
                        outb[(size_t)gm * DTRP + gn] =
                            (gn < DTR) ? f2bf(v) : f2bf(0.f);
                } else if (MODE == 3) {
                    float s  = v + bias[gn];
                    float sp = (s > 20.f) ? s : log1pf(__expf(s));
                    out0[(size_t)gm * N + gn] = sp;
                } else if (MODE == 4) {
                    out0[(size_t)gm * N + gn] = resid[(size_t)gm * N + gn] + v;
                } else {
                    out0[(size_t)gm * N + gn] = v;
                }
            }
        }
    }
}

// ---------------- depthwise causal conv (K=4) + bias + SiLU ----------------
__global__ __launch_bounds__(256) void conv_silu_kernel(
    const float* __restrict__ xc, const float* __restrict__ w,
    const float* __restrict__ cb, float* __restrict__ of, bf16* __restrict__ ob)
{
    const int idx = blockIdx.x * 256 + threadIdx.x;
    if (idx >= ROWS * DIc) return;
    const int d   = idx % DIc;
    const int row = idx / DIc;
    const int l   = row % Ll;
    float acc = cb[d];
#pragma unroll
    for (int j = 0; j < Kc; ++j) {
        const int ll = l - (Kc - 1) + j;
        if (ll >= 0)
            acc += w[d * Kc + j] * xc[(size_t)(row - (Kc - 1) + j) * DIc + d];
    }
    const float s = acc / (1.f + __expf(-acc));     // SiLU
    of[idx] = s;
    ob[idx] = f2bf(s);
}

// ---------------- selective scan (sequential over L, parallel over B*DI) ----
__global__ __launch_bounds__(256) void scan_kernel(
    const float* __restrict__ dt, const float* __restrict__ dbl,
    const float* __restrict__ xc, const float* __restrict__ z,
    const float* __restrict__ A_log, const float* __restrict__ Dp,
    bf16* __restrict__ yb)
{
    const int t = blockIdx.x * 256 + threadIdx.x;
    if (t >= Bb * DIc) return;
    const int b = t / DIc, d = t % DIc;

    float Aa[Nn], h[Nn];
#pragma unroll
    for (int n = 0; n < Nn; ++n) { Aa[n] = -__expf(A_log[d * Nn + n]); h[n] = 0.f; }
    const float dpar = Dp[d];

    for (int l = 0; l < Ll; ++l) {
        const size_t row = (size_t)b * Ll + l;
        const float dtv = dt[row * DIc + d];
        const float xv  = xc[row * DIc + d];
        const float zv  = z [row * DIc + d];
        const float* bc = dbl + row * XPROJ;
        const float dtx = dtv * xv;
        float y = 0.f;
#pragma unroll
        for (int n = 0; n < Nn; ++n) {
            h[n] = __expf(dtv * Aa[n]) * h[n] + dtx * bc[DTR + n];
            y   += h[n] * bc[DTR + Nn + n];
        }
        y += xv * dpar;
        y *= zv / (1.f + __expf(-zv));              // * SiLU(z)
        yb[row * DIc + d] = f2bf(y);
    }
}

// ---------------- host: build pipeline on stream ----------------
extern "C" void kernel_launch(void* const* d_in, const int* in_sizes, int n_in,
                              void* d_out, int out_size, void* d_ws, size_t ws_size,
                              hipStream_t stream)
{
    (void)in_sizes; (void)n_in; (void)out_size; (void)ws_size;
    const float* x    = (const float*)d_in[0];
    const float* g    = (const float*)d_in[1];
    const float* be   = (const float*)d_in[2];
    const float* Win  = (const float*)d_in[3];
    const float* cw   = (const float*)d_in[4];
    const float* cb   = (const float*)d_in[5];
    const float* Wx   = (const float*)d_in[6];
    const float* Wdt  = (const float*)d_in[7];
    const float* bdt  = (const float*)d_in[8];
    const float* Alog = (const float*)d_in[9];
    const float* Dp   = (const float*)d_in[10];
    const float* Wout = (const float*)d_in[11];
    float* out = (float*)d_out;

    char* ws = (char*)d_ws;
    size_t off = 0;
    auto alloc = [&](size_t bytes) -> char* {
        char* p = ws + off;
        off = (off + bytes + 255) & ~(size_t)255;
        return p;
    };
    bf16*  xn_bf   = (bf16*) alloc((size_t)ROWS * Dd * 2);
    bf16*  Win_bf  = (bf16*) alloc((size_t)2 * DIc * Dd * 2);
    float* xc_f    = (float*)alloc((size_t)ROWS * DIc * 4);  // reused as dt buffer
    float* z_f     = (float*)alloc((size_t)ROWS * DIc * 4);
    float* xcs_f   = (float*)alloc((size_t)ROWS * DIc * 4);
    bf16*  xcs_bf  = (bf16*) alloc((size_t)ROWS * DIc * 2);
    bf16*  Wx_bf   = (bf16*) alloc((size_t)XPROJ * DIc * 2);
    float* dbl_f   = (float*)alloc((size_t)ROWS * XPROJ * 4);
    bf16*  dtr_bf  = (bf16*) alloc((size_t)ROWS * DTRP * 2);   // K padded to 64
    bf16*  Wdt_bf  = (bf16*) alloc((size_t)DIc * DTRP * 2);    // K padded to 64
    bf16*  y_bf    = (bf16*) alloc((size_t)ROWS * DIc * 2);
    bf16*  Wout_bf = (bf16*) alloc((size_t)Dd * DIc * 2);

    // 1) LayerNorm -> bf16 activations
    layernorm_kernel<<<ROWS, 256, 0, stream>>>(x, g, be, xn_bf);

    // 2) weight converts to bf16 (tiny, amortized)
    int n;
    n = 2 * DIc * Dd;  convert_bf16_kernel<<<(n + 255) / 256, 256, 0, stream>>>(Win,  Win_bf,  n);
    n = XPROJ * DIc;   convert_bf16_kernel<<<(n + 255) / 256, 256, 0, stream>>>(Wx,   Wx_bf,   n);
    n = Dd * DIc;      convert_bf16_kernel<<<(n + 255) / 256, 256, 0, stream>>>(Wout, Wout_bf, n);
    n = DIc * DTRP;    convert_pad_bf16_kernel<<<(n + 255) / 256, 256, 0, stream>>>(
                           Wdt, Wdt_bf, DIc, DTR, DTRP);

    // 3) in_proj: (4096x768)x(3072x768)^T, split into xc | z
    wmma_gemm_kernel<1, true><<<dim3(ROWS / 256, (2 * DIc) / 64), 256, 0, stream>>>(
        xn_bf, Win_bf, 2 * DIc, Dd, xc_f, z_f, nullptr, nullptr, nullptr);

    // 4) depthwise causal conv + SiLU
    conv_silu_kernel<<<(ROWS * DIc + 255) / 256, 256, 0, stream>>>(xc_f, cw, cb, xcs_f, xcs_bf);

    // 5) x_proj: (4096x1536)x(80x1536)^T -> dbl (f32) + padded bf16 dt-lowrank
    wmma_gemm_kernel<2, false><<<dim3(ROWS / 256, (XPROJ + 63) / 64), 256, 0, stream>>>(
        xcs_bf, Wx_bf, XPROJ, DIc, dbl_f, nullptr, dtr_bf, nullptr, nullptr);

    // 6) dt_proj + softplus: (4096x64)x(1536x64)^T -> dt (reuses xc_f)
    wmma_gemm_kernel<3, true><<<dim3(ROWS / 256, DIc / 64), 256, 0, stream>>>(
        dtr_bf, Wdt_bf, DIc, DTRP, xc_f, nullptr, nullptr, bdt, nullptr);

    // 7) selective scan + D skip + SiLU(z) gate -> bf16 y
    scan_kernel<<<(Bb * DIc + 255) / 256, 256, 0, stream>>>(
        xc_f, dbl_f, xcs_f, z_f, Alog, Dp, y_bf);

    // 8) out_proj + residual: (4096x1536)x(768x1536)^T + x -> out
    wmma_gemm_kernel<4, true><<<dim3(ROWS / 256, Dd / 64), 256, 0, stream>>>(
        y_bf, Wout_bf, Dd, DIc, out, nullptr, nullptr, nullptr, x);
}
